// PG_Explainer_781684048328
// MI455X (gfx1250) — matrix operand order repaired and checked by reading, not compiled, not run on hardware
//
#include <hip/hip_runtime.h>
#include <hip/hip_bf16.h>
#include <math.h>

// ---- CDNA5 WMMA vector types ----
typedef __attribute__((ext_vector_type(16))) _Float16 v16h;
typedef __attribute__((ext_vector_type(8)))  _Float16 v8h;
typedef __attribute__((ext_vector_type(8)))  float    v8f;

#define N_NODES 1024
#define HID 25
#define MLP_H 64
#define N_CLS 7
#define NPAD 32      // xw columns padded 25 -> 32 for WMMA
#define LSTR 68      // LDS row stride (dwords): 16B-aligned rows, conflict-free

// ------------------------------------------------------------------
// Kernel 1: prep
//  Afold[i][h] = b1[h] + sum_k embed[i][k] * W1[k][h]          (h<64)
//  Bv[i][h]   =          sum_k embed[i][k] * W1[25+k][h]
//  XWT[n][i]  = f16( sum_k x[i][k] * Wm1[k][n] )  (n<25, else 0) -- transposed,
//               K-contiguous so WMMA B fragments are contiguous loads.
// ------------------------------------------------------------------
__global__ void pg_prep_kernel(const float* __restrict__ x,
                               const float* __restrict__ embed,
                               const float* __restrict__ W1,
                               const float* __restrict__ b1,
                               const float* __restrict__ Wm1,
                               float* __restrict__ Afold,
                               float* __restrict__ Bv,
                               _Float16* __restrict__ XWT) {
  const int i = blockIdx.x;       // node row
  const int t = threadIdx.x;      // 64 threads
  __shared__ float erow[HID];
  __shared__ float xrow[MLP_H];
  if (t < HID) erow[t] = embed[i * HID + t];
  xrow[t] = x[i * MLP_H + t];
  __syncthreads();

  float sa = b1[t], sb = 0.f;
#pragma unroll 5
  for (int k = 0; k < HID; ++k) {
    const float e = erow[k];
    sa = fmaf(e, W1[k * MLP_H + t], sa);            // W1[:25] rows
    sb = fmaf(e, W1[(HID + k) * MLP_H + t], sb);    // W1[25:] rows
  }
  Afold[i * MLP_H + t] = sa;
  Bv[i * MLP_H + t]    = sb;

  if (t < NPAD) {
    float v = 0.f;
    if (t < HID) {
#pragma unroll 8
      for (int k = 0; k < MLP_H; ++k) v = fmaf(xrow[k], Wm1[k * HID + t], v);
    }
    XWT[t * N_NODES + i] = (_Float16)v;             // zero for padded cols
  }
}

// ------------------------------------------------------------------
// Kernel 2: mask  (the VALU-dominant 1M-pair loop, triangular schedule)
// Block (I,J), J>=I, 512 threads = two 256-thread halves:
//   half 0: thread (r,c) computes gate(I16+r, J16+c)   -> tile (I,J)
//   half 1: thread (r,c) computes gate(J16+r, I16+c)   -> tile (J,I)
// Inner loop reads float4 chunks from LDS (ds_load_b128): a & W2 are lane
// broadcasts, b is bank-conflict-free (stride 68 dwords).
// sym = 0.5*(g + g^T) via LDS transpose; each half stores its own tile of
// masked_adj = adj * sym as f16 (diagonal tile stored once).
// ------------------------------------------------------------------
__global__ void pg_mask_kernel(const float* __restrict__ Afold,
                               const float* __restrict__ Bv,
                               const float* __restrict__ adj,
                               const float* __restrict__ noise,
                               const float* __restrict__ tmp,
                               const float* __restrict__ b2,
                               const float* __restrict__ W2,
                               _Float16* __restrict__ MA) {
  const int I = blockIdx.x, J = blockIdx.y;
  if (J < I) return;                                  // triangular schedule

  __shared__ float sA[2][16 * LSTR];  // [0]=Afold I-rows, [1]=Afold J-rows
  __shared__ float sB[2][16 * LSTR];  // [0]=Bv    J-rows, [1]=Bv    I-rows
  __shared__ float W2s[MLP_H];
  __shared__ float g[2][16 * 17];

  const int t = threadIdx.x;          // 512
  for (int idx = t; idx < 16 * MLP_H; idx += 512) {
    const int row = idx >> 6, col = idx & 63;
    sA[0][row * LSTR + col] = Afold[(I * 16 + row) * MLP_H + col];
    sA[1][row * LSTR + col] = Afold[(J * 16 + row) * MLP_H + col];
    sB[0][row * LSTR + col] = Bv   [(J * 16 + row) * MLP_H + col];
    sB[1][row * LSTR + col] = Bv   [(I * 16 + row) * MLP_H + col];
  }
  if (t < MLP_H) W2s[t] = W2[t];
  __syncthreads();

  const int half = t >> 8;            // 0 or 1
  const int tt = t & 255;
  const int r = tt >> 4, c = tt & 15;

  const float* __restrict__ arow = &sA[half][r * LSTR];
  const float* __restrict__ brow = &sB[half][c * LSTR];

  float acc = 0.f;
#pragma unroll
  for (int hh = 0; hh < MLP_H / 4; ++hh) {
    const float4 a4 = *(const float4*)(arow + 4 * hh);   // ds_load_b128 (bcast)
    const float4 b4 = *(const float4*)(brow + 4 * hh);   // ds_load_b128
    const float4 w4 = *(const float4*)(&W2s[4 * hh]);    // ds_load_b128 (bcast)
    acc = fmaf(fmaxf(a4.x + b4.x, 0.f), w4.x, acc);
    acc = fmaf(fmaxf(a4.y + b4.y, 0.f), w4.y, acc);
    acc = fmaf(fmaxf(a4.z + b4.z, 0.f), w4.z, acc);
    acc = fmaf(fmaxf(a4.w + b4.w, 0.f), w4.w, acc);
  }
  const float la = fmaxf(acc + b2[0], 0.f);

  // global (row,col) for this thread's gate
  const int gr = (half ? J : I) * 16 + r;
  const int gc = (half ? I : J) * 16 + c;
  const int gi = gr * N_NODES + gc;

  const float u  = noise[gi];
  const float lg = __logf(u) - log1pf(-u);
  const float invt = 1.0f / tmp[0];
  const float gate = 1.f / (1.f + __expf(-(lg + la) * invt));

  g[half][r * 17 + c] = gate;
  __syncthreads();

  // sym_mask(gr,gc) = 0.5*(gate(gr,gc) + gate(gc,gr))
  const float symv = 0.5f * (g[half][r * 17 + c] + g[1 - half][c * 17 + r]);
  if (half == 0 || I != J)                    // diagonal tile: store once
    MA[gi] = (_Float16)(adj[gi] * symv);
}

// ------------------------------------------------------------------
// Kernel 3: WMMA GEMM  hm = relu( MA(f16)[1024,1024] @ xw[1024,32] )
// 64 blocks (one 16-row M tile each) x 8 waves; wave w reduces K slice
// [w*128, w*128+128) via v_wmma_f32_16x16x32_f16, cross-wave sum in LDS.
// A fragment (16x32 f16, row-major MA): lane m<16 -> row m, K k0..k0+15;
// lane 16+m -> K k0+16..k0+31 (matches B's K order).
// B fragment (32x16 f16) from XWT (N-major, K-contiguous): lane n<16 ->
// col n, K k0..k0+15; lane 16+n -> col n, K k0+16..k0+31.
// ------------------------------------------------------------------
__global__ void pg_gemm_kernel(const _Float16* __restrict__ MA,
                               const _Float16* __restrict__ XWT,
                               float* __restrict__ hm) {
  __shared__ float red[8][16][NPAD];   // 16 KB
  const int t = threadIdx.x;           // 256
  const int wave = t >> 5;
  const int lane = t & 31;
  const int laneHi = lane >> 4;        // 0 or 1
  const int laneLo = lane & 15;
  const int m0 = blockIdx.x * 16;
  const int row = m0 + laneLo;

  v8f c0 = {};
  v8f c1 = {};
  const int kstart = wave * 128;
#pragma unroll
  for (int ks = 0; ks < 4; ++ks) {
    const int k0 = kstart + ks * 32;
    // ---- A fragment ----
    const int kbase = k0 + laneHi * 16;
    const _Float16* pa = MA + row * N_NODES + kbase;
    const v8h alo = *(const v8h*)(pa);
    const v8h ahi = *(const v8h*)(pa + 8);
    v16h afrag;
#pragma unroll
    for (int e = 0; e < 8; ++e) { afrag[e] = alo[e]; afrag[8 + e] = ahi[e]; }
    // ---- B fragments (two N halves) ----
    const int koff = k0 + laneHi * 16;
    const v16h b0 = *(const v16h*)(XWT + laneLo * N_NODES + koff);
    const v16h b1 = *(const v16h*)(XWT + (16 + laneLo) * N_NODES + koff);
    c0 = __builtin_amdgcn_wmma_f32_16x16x32_f16(false, afrag, false, b0,
                                                (short)0, c0, false, false);
    c1 = __builtin_amdgcn_wmma_f32_16x16x32_f16(false, afrag, false, b1,
                                                (short)0, c1, false, false);
  }
  // C layout: element e -> (M = e + 8*laneHi, N = laneLo [+16 for c1])
#pragma unroll
  for (int e = 0; e < 8; ++e) {
    const int M = e + 8 * laneHi;
    red[wave][M][laneLo]      = c0[e];
    red[wave][M][16 + laneLo] = c1[e];
  }
  __syncthreads();
  for (int o = t; o < 16 * NPAD; o += 256) {
    const int M = o >> 5, Nn = o & 31;
    float s = 0.f;
#pragma unroll
    for (int w = 0; w < 8; ++w) s += red[w][M][Nn];
    hm[(m0 + M) * NPAD + Nn] = fmaxf(s, 0.f);       // ReLU after full K sum
  }
}

// ------------------------------------------------------------------
// Kernel 4: column mean over 1024 rows, @ Wm2 [25,7], softmax -> out[7]
// ------------------------------------------------------------------
__global__ void pg_finish_kernel(const float* __restrict__ hm,
                                 const float* __restrict__ Wm2,
                                 float* __restrict__ out) {
  __shared__ float redsum[8][NPAD];
  __shared__ float colmean[NPAD];
  const int t = threadIdx.x;           // 256
  const int col = t & 31, g = t >> 5;
  float s = 0.f;
  for (int r = g; r < N_NODES; r += 8) s += hm[r * NPAD + col];
  redsum[g][col] = s;
  __syncthreads();
  if (t < NPAD) {
    float tot = 0.f;
#pragma unroll
    for (int gg = 0; gg < 8; ++gg) tot += redsum[gg][t];
    colmean[t] = tot * (1.0f / (float)N_NODES);
  }
  __syncthreads();
  if (t == 0) {
    float logits[N_CLS];
    float m = -1e30f;
    for (int c = 0; c < N_CLS; ++c) {
      float l = 0.f;
      for (int h = 0; h < HID; ++h) l += colmean[h] * Wm2[h * N_CLS + c];
      logits[c] = l;
      m = fmaxf(m, l);
    }
    float sum = 0.f;
    for (int c = 0; c < N_CLS; ++c) { logits[c] = __expf(logits[c] - m); sum += logits[c]; }
    for (int c = 0; c < N_CLS; ++c) out[c] = logits[c] / sum;
  }
}

// ------------------------------------------------------------------
// Launcher.  Workspace layout (bytes), all regions 256B aligned:
//   [0,        2097152) MA    f16 [1024][1024]
//   [2097152,  2359296) Afold f32 [1024][64]
//   [2359296,  2621440) Bv    f32 [1024][64]
//   [2621440,  2686976) XWT   f16 [32][1024]
//   [2686976,  2818048) hm    f32 [1024][32]
// ------------------------------------------------------------------
extern "C" void kernel_launch(void* const* d_in, const int* in_sizes, int n_in,
                              void* d_out, int out_size, void* d_ws, size_t ws_size,
                              hipStream_t stream) {
  const float* x     = (const float*)d_in[0];
  const float* embed = (const float*)d_in[1];
  const float* adj   = (const float*)d_in[2];
  const float* tmp   = (const float*)d_in[3];
  /* d_in[4] = label: unused by the reference output */
  const float* noise = (const float*)d_in[5];
  const float* W1    = (const float*)d_in[6];
  const float* b1    = (const float*)d_in[7];
  const float* W2    = (const float*)d_in[8];
  const float* b2    = (const float*)d_in[9];
  const float* Wm1   = (const float*)d_in[10];
  const float* Wm2   = (const float*)d_in[11];
  float* out = (float*)d_out;

  char* ws = (char*)d_ws;
  _Float16* MA    = (_Float16*)(ws);
  float*    Afold = (float*)(ws + 2097152);
  float*    Bv    = (float*)(ws + 2359296);
  _Float16* XWT   = (_Float16*)(ws + 2621440);
  float*    hm    = (float*)(ws + 2686976);

  pg_prep_kernel<<<N_NODES, 64, 0, stream>>>(x, embed, W1, b1, Wm1, Afold, Bv, XWT);

  dim3 grid_mask(N_NODES / 16, N_NODES / 16);        // triangular: J<I exits
  pg_mask_kernel<<<grid_mask, 512, 0, stream>>>(Afold, Bv, adj, noise, tmp, b2, W2, MA);

  pg_gemm_kernel<<<N_NODES / 16, 256, 0, stream>>>(MA, XWT, hm);

  pg_finish_kernel<<<1, 256, 0, stream>>>(hm, Wm2, out);
}
